// MambaVisionMixer_35983236006215
// MI455X (gfx1250) — matrix operand order repaired and checked by reading, not compiled, not run on hardware
//
#include <hip/hip_runtime.h>
#include <hip/hip_bf16.h>

// ---------------------------------------------------------------------------
// MambaVision mixer for MI455X (gfx1250, wave32, WMMA bf16 + async-LDS path)
// ---------------------------------------------------------------------------

namespace {
constexpr int kB    = 8;
constexpr int kL    = 4097;
constexpr int kDM   = 384;   // d_model
constexpr int kDI   = 768;   // d_inner
constexpr int kDH   = 384;   // d_half
constexpr int kNS   = 16;    // d_state
constexpr int kDTR  = 24;    // dt_rank
constexpr int kNDBL = 56;    // dt_rank + 2*d_state
constexpr int kTT   = (kL + 15) / 16;  // 257 token tiles
}

typedef __bf16 v16bf __attribute__((ext_vector_type(16)));
typedef float  v8f   __attribute__((ext_vector_type(8)));

// Async global->LDS builtin operand types (from hipcc diagnostic: the builtin
// takes pointers to int __attribute__((vector_size(16))) in AS1 / AS3).
typedef __attribute__((__vector_size__(4 * sizeof(int)))) int i128v;
typedef __attribute__((address_space(1))) i128v* i128v_g;
typedef __attribute__((address_space(3))) i128v* i128v_l;

#if defined(__has_builtin)
#if __has_builtin(__builtin_amdgcn_global_load_async_to_lds_b128)
#define USE_ASYNC_LDS 1
#endif
#endif

union Frag {
  v16bf          v;
  __bf16         h[16];
  unsigned int   u[8];
  unsigned short us[16];
};

// Pair-base K offset for the 16-bit A fragment (contiguous K order, matches B).
// Pair p holds elements (2p, 2p+1); lanes 16..31 are offset by K+16.
__device__ __forceinline__ int kpairA(int hi, int p) {
  return (p << 1) + (hi << 4);
}

__device__ __forceinline__ v8f wmma_bf16(const Frag& a, const Frag& b, v8f c) {
  return __builtin_amdgcn_wmma_f32_16x16x32_bf16(
      false, a.v, false, b.v, (short)0, c, false, false);
}

// Stage one 16-token x 384-K bf16 activation strip into LDS (12 KB).
// 128 threads, 16B chunks; uses gfx1250 async global->LDS copies when the
// toolchain exposes them, plain LDS stores otherwise.
__device__ __forceinline__ void stage_token_tile(
    const __bf16* __restrict__ act, __bf16* smem, int b, int t0, int tid) {
  const int chunks = 16 * kDM / 8;          // 768 x 16-byte chunks
  for (int c = tid; c < chunks; c += 128) {
    int row = c / 48;                       // token within tile
    int col = c % 48;                       // 16B chunk within row
    int tok = min(t0 + row, kL - 1);
    const __bf16* g = act + ((size_t)b * kL + tok) * kDM + col * 8;
    __bf16* s = smem + row * kDM + col * 8;
#ifdef USE_ASYNC_LDS
    __builtin_amdgcn_global_load_async_to_lds_b128((i128v_g)g, (i128v_l)s, 0, 0);
#else
    *(uint4*)s = *(const uint4*)g;
#endif
  }
#ifdef USE_ASYNC_LDS
#if __has_builtin(__builtin_amdgcn_s_wait_asynccnt)
  __builtin_amdgcn_s_wait_asynccnt(0);
#else
  asm volatile("s_wait_asynccnt 0x0" ::: "memory");
#endif
#endif
  __syncthreads();
}

// ---------------------------------------------------------------------------
// Prep: fp32 -> bf16 weight conversion
// ---------------------------------------------------------------------------
__global__ void cvt_bf16_kernel(const float* __restrict__ src,
                                __bf16* __restrict__ dst, int n) {
  int i = blockIdx.x * blockDim.x + threadIdx.x;
  if (i < n) dst[i] = (__bf16)src[i];
}

// dt_proj_w (384x24) -> bf16 padded to (384x32) with zeros
__global__ void cvt_dtw_kernel(const float* __restrict__ src,
                               __bf16* __restrict__ dst) {
  int i = blockIdx.x * blockDim.x + threadIdx.x;
  if (i >= kDH * 32) return;
  int r = i >> 5, k = i & 31;
  dst[i] = (k < kDTR) ? (__bf16)src[r * kDTR + k] : (__bf16)0.f;
}

// hidden_states gather (apply token permutation) + bf16 convert.
// dst layout: (B, L, 384) row-major, tokens already in scan order.
__global__ void gather_hs_kernel(const float* __restrict__ hs,
                                 const long long* __restrict__ perm,
                                 __bf16* __restrict__ dst) {
  size_t i = (size_t)blockIdx.x * blockDim.x + threadIdx.x;
  const size_t total = (size_t)kB * kL * kDM;
  if (i >= total) return;
  int k = (int)(i % kDM);
  size_t t = i / kDM;
  int l = (int)(t % kL);
  int b = (int)(t / kL);
  int src_l = (l == 0) ? 0 : (int)(1 + perm[l - 1]);
  dst[i] = (__bf16)hs[((size_t)b * kL + src_l) * kDM + k];
}

// ---------------------------------------------------------------------------
// GEMM 1: xz[b, e, l] = sum_k w_in[e,k] * hs_perm[b,l,k]
// 4 waves/WG share one LDS token strip; each wave register-blocks 4 channel
// subtiles (64 channels) -> 256 channels per WG, grid.x = 3.
// ---------------------------------------------------------------------------
__global__ __launch_bounds__(128) void gemm_in_proj_kernel(
    const __bf16* __restrict__ act,   // (B, L, 384) bf16, permuted tokens
    const __bf16* __restrict__ w,     // (768, 384) bf16
    float* __restrict__ xz)           // (B, 768, L) f32
{
  __shared__ __bf16 smem[16 * kDM];
  const int tid  = threadIdx.x;
  const int lane = tid & 31;
  const int wv   = tid >> 5;
  const int lo = lane & 15, hi = lane >> 4;
  const int b  = blockIdx.z;
  const int t0 = blockIdx.y * 16;
  const int e0 = blockIdx.x * 256 + wv * 64;

  stage_token_tile(act, smem, b, t0, tid);

  const __bf16* arow = w + (size_t)(e0 + lo) * kDM;
  const __bf16* srow = smem + lo * kDM;

  v8f acc[4] = {};
  for (int kc = 0; kc < kDM; kc += 32) {
    Frag bfr;
#pragma unroll
    for (int p = 0; p < 8; ++p)
      bfr.u[p] = *(const unsigned int*)(srow + kc + 2 * p + 16 * hi);
    Frag a0, a1, a2, a3;
#pragma unroll
    for (int p = 0; p < 8; ++p) {
      int ko = kc + kpairA(hi, p);
      a0.u[p] = *(const unsigned int*)(arow + ko);
      a1.u[p] = *(const unsigned int*)(arow + 16 * kDM + ko);
      a2.u[p] = *(const unsigned int*)(arow + 32 * kDM + ko);
      a3.u[p] = *(const unsigned int*)(arow + 48 * kDM + ko);
    }
    acc[0] = wmma_bf16(a0, bfr, acc[0]);
    acc[1] = wmma_bf16(a1, bfr, acc[1]);
    acc[2] = wmma_bf16(a2, bfr, acc[2]);
    acc[3] = wmma_bf16(a3, bfr, acc[3]);
  }
  const int t = t0 + lo;
  if (t < kL) {
#pragma unroll
    for (int s = 0; s < 4; ++s)
#pragma unroll
      for (int r = 0; r < 8; ++r) {
        int ch = e0 + s * 16 + r + 8 * hi;     // C/D layout: M=r (+8 hi-half)
        xz[((size_t)b * kDI + ch) * kL + t] = acc[s][r];
      }
  }
}

// ---------------------------------------------------------------------------
// Depthwise conv (K=4, pad left=1/right=2) + SiLU; also emit bf16 copy of x
// in (B, L, 384) layout for the x_proj WMMA GEMM.
// ---------------------------------------------------------------------------
__global__ void conv_silu_kernel(const float* __restrict__ xz,
                                 const float* __restrict__ wx,
                                 const float* __restrict__ wz,
                                 float* __restrict__ xzc,
                                 __bf16* __restrict__ xbf) {
  size_t i = (size_t)blockIdx.x * blockDim.x + threadIdx.x;
  const size_t total = (size_t)kB * kDI * kL;
  if (i >= total) return;
  int l = (int)(i % kL);
  size_t t = i / kL;
  int c = (int)(t % kDI);
  int b = (int)(t / kDI);
  const float* wp = (c < kDH) ? (wx + c * 4) : (wz + (c - kDH) * 4);
  const float* row = xz + ((size_t)b * kDI + c) * kL;
  float acc = 0.f;
#pragma unroll
  for (int k = 0; k < 4; ++k) {
    int src = l + k - 1;
    float v = (src >= 0 && src < kL) ? row[src] : 0.f;
    acc += wp[k] * v;
  }
  float s = acc / (1.f + __expf(-acc));   // SiLU
  xzc[i] = s;
  if (c < kDH) xbf[((size_t)b * kL + l) * kDH + c] = (__bf16)s;
}

// ---------------------------------------------------------------------------
// GEMM 2: x_dbl[b, e, l] = sum_k x_proj_w[e,k] * x[b,l,k]
// One wave covers all 4 row-subtiles (56 rows padded to 64) from a single
// shared B fragment -> activations read once.
// ---------------------------------------------------------------------------
__global__ __launch_bounds__(32) void gemm_x_proj_kernel(
    const __bf16* __restrict__ act,   // xbf (B, L, 384)
    const __bf16* __restrict__ w,     // (56, 384) bf16
    float* __restrict__ xdbl)         // (B, 56, L) f32
{
  const int lane = threadIdx.x;
  const int lo = lane & 15, hi = lane >> 4;
  const int b  = blockIdx.z;
  const int t0 = blockIdx.y * 16;
  const int tokc = min(t0 + lo, kL - 1);

  const __bf16* brow = act + ((size_t)b * kL + tokc) * kDH;
  const __bf16* a0r = w + (size_t)min(0 * 16 + lo, kNDBL - 1) * kDM;
  const __bf16* a1r = w + (size_t)min(1 * 16 + lo, kNDBL - 1) * kDM;
  const __bf16* a2r = w + (size_t)min(2 * 16 + lo, kNDBL - 1) * kDM;
  const __bf16* a3r = w + (size_t)min(3 * 16 + lo, kNDBL - 1) * kDM;

  v8f acc[4] = {};
  for (int kc = 0; kc < kDM; kc += 32) {
    Frag bfr;
#pragma unroll
    for (int p = 0; p < 8; ++p)
      bfr.u[p] = *(const unsigned int*)(brow + kc + 2 * p + 16 * hi);
    Frag a0, a1, a2, a3;
#pragma unroll
    for (int p = 0; p < 8; ++p) {
      int ko = kc + kpairA(hi, p);
      a0.u[p] = *(const unsigned int*)(a0r + ko);
      a1.u[p] = *(const unsigned int*)(a1r + ko);
      a2.u[p] = *(const unsigned int*)(a2r + ko);
      a3.u[p] = *(const unsigned int*)(a3r + ko);
    }
    acc[0] = wmma_bf16(a0, bfr, acc[0]);
    acc[1] = wmma_bf16(a1, bfr, acc[1]);
    acc[2] = wmma_bf16(a2, bfr, acc[2]);
    acc[3] = wmma_bf16(a3, bfr, acc[3]);
  }
  const int t = t0 + lo;
  if (t < kL) {
#pragma unroll
    for (int s = 0; s < 4; ++s)
#pragma unroll
      for (int r = 0; r < 8; ++r) {
        int ch = s * 16 + r + 8 * hi;
        if (ch < kNDBL) xdbl[((size_t)b * kNDBL + ch) * kL + t] = acc[s][r];
      }
  }
}

// ---------------------------------------------------------------------------
// GEMM 3: dt[b, d, l] = sum_r dt_proj_w[d,r] * dt_low[b,r,l]  (K=24 -> 32pad)
// ---------------------------------------------------------------------------
__global__ __launch_bounds__(32) void gemm_dt_proj_kernel(
    const float* __restrict__ xdbl,   // (B, 56, L)
    const __bf16* __restrict__ wdt,   // (384, 32) bf16, zero-padded
    float* __restrict__ dt)           // (B, 384, L)
{
  const int lane = threadIdx.x;
  const int lo = lane & 15, hi = lane >> 4;
  const int b  = blockIdx.z;
  const int t0 = blockIdx.y * 16;
  const int e0 = blockIdx.x * 16;
  const int tokc = min(t0 + lo, kL - 1);

  Frag a, bf;
#pragma unroll
  for (int p = 0; p < 8; ++p)
    a.u[p] = *(const unsigned int*)(wdt + (size_t)(e0 + lo) * 32 + kpairA(hi, p));
#pragma unroll
  for (int p = 0; p < 8; ++p) {
    int k0 = 2 * p + 16 * hi;
    float f0 = (k0     < kDTR) ? xdbl[((size_t)b * kNDBL + k0    ) * kL + tokc] : 0.f;
    float f1 = (k0 + 1 < kDTR) ? xdbl[((size_t)b * kNDBL + k0 + 1) * kL + tokc] : 0.f;
    bf.h[2 * p]     = (__bf16)f0;
    bf.h[2 * p + 1] = (__bf16)f1;
  }
  v8f acc = {};
  acc = wmma_bf16(a, bf, acc);

  const int t = t0 + lo;
  if (t < kL) {
#pragma unroll
    for (int r = 0; r < 8; ++r) {
      int ch = e0 + r + 8 * hi;
      dt[((size_t)b * kDH + ch) * kL + t] = acc[r];
    }
  }
}

// ---------------------------------------------------------------------------
// Selective scan: one lane per (b, d, n); 16-lane shfl_xor tree reduces h.C.
// Fuses softplus(dt+bias), D*x skip, z-gating; emits y in bf16 (B, L, 384).
// ---------------------------------------------------------------------------
__global__ __launch_bounds__(256) void scan_kernel(
    const float* __restrict__ dt,     // (B, 384, L)
    const float* __restrict__ xzc,    // (B, 768, L)  x | z (post conv+silu)
    const float* __restrict__ xdbl,   // (B, 56, L)   rows 24..39=B, 40..55=C
    const float* __restrict__ A_log,  // (384, 16)
    const float* __restrict__ dtb,    // (384,)
    const float* __restrict__ Dp,     // (384,)
    __bf16* __restrict__ ybf)         // (B, L, 384)
{
  const int n = threadIdx.x & 15;
  const int d = blockIdx.x * 16 + (threadIdx.x >> 4);
  const int b = blockIdx.y;

  const float A_dn = -__expf(A_log[d * kNS + n]);
  const float bias = dtb[d];
  const float Dv   = Dp[d];

  const float* dtp = dt   + ((size_t)b * kDH + d) * kL;
  const float* xp  = xzc  + ((size_t)b * kDI + d) * kL;
  const float* zp  = xzc  + ((size_t)b * kDI + kDH + d) * kL;
  const float* Bp  = xdbl + ((size_t)b * kNDBL + kDTR + n) * kL;
  const float* Cp  = xdbl + ((size_t)b * kNDBL + kDTR + kNS + n) * kL;
  __bf16* yrow = ybf + (size_t)b * kL * kDH + d;

  float h = 0.f;
  for (int l = 0; l < kL; ++l) {
    if ((l & 63) == 0 && (l + 256) < kL) {
      __builtin_prefetch(dtp + l + 256, 0, 1);   // global_prefetch_b8
      __builtin_prefetch(xp + l + 256, 0, 1);
      __builtin_prefetch(zp + l + 256, 0, 1);
    }
    float dv = dtp[l] + bias;
    dv = (dv > 20.f) ? dv : log1pf(__expf(dv));  // softplus
    float xv = xp[l];
    float dA = __expf(dv * A_dn);
    h = h * dA + (dv * xv) * Bp[l];
    float p = h * Cp[l];
    p += __shfl_xor(p, 1, 16);
    p += __shfl_xor(p, 2, 16);
    p += __shfl_xor(p, 4, 16);
    p += __shfl_xor(p, 8, 16);
    float y = (p + Dv * xv) * zp[l];
    if (n == 0) yrow[(size_t)l * kDH] = (__bf16)y;
  }
}

// ---------------------------------------------------------------------------
// GEMM 4: out[b, dst(l), e] = sum_d out_proj_w[e,d] * y[b,l,d]
// 4 waves share an LDS token strip; each wave does 3 channel subtiles
// (48 ch) -> 192 channels per WG, grid.x = 3. Inverse permutation folded
// into the scatter store (dst = 1 + perm[l-1]).
// ---------------------------------------------------------------------------
__global__ __launch_bounds__(128) void gemm_out_proj_kernel(
    const __bf16* __restrict__ act,   // ybf (B, L, 384)
    const __bf16* __restrict__ w,     // (384, 384) bf16
    const long long* __restrict__ perm,
    float* __restrict__ out)          // (B, L, 384)
{
  __shared__ __bf16 smem[16 * kDH];
  const int tid  = threadIdx.x;
  const int lane = tid & 31;
  const int wv   = tid >> 5;
  const int lo = lane & 15, hi = lane >> 4;
  const int b  = blockIdx.z;
  const int t0 = blockIdx.y * 16;
  const int e0 = blockIdx.x * 192 + wv * 48;

  stage_token_tile(act, smem, b, t0, tid);

  const __bf16* arow = w + (size_t)(e0 + lo) * kDH;
  const __bf16* srow = smem + lo * kDH;

  v8f acc[3] = {};
  for (int kc = 0; kc < kDH; kc += 32) {
    Frag bfr;
#pragma unroll
    for (int p = 0; p < 8; ++p)
      bfr.u[p] = *(const unsigned int*)(srow + kc + 2 * p + 16 * hi);
    Frag a0, a1, a2;
#pragma unroll
    for (int p = 0; p < 8; ++p) {
      int ko = kc + kpairA(hi, p);
      a0.u[p] = *(const unsigned int*)(arow + ko);
      a1.u[p] = *(const unsigned int*)(arow + 16 * kDH + ko);
      a2.u[p] = *(const unsigned int*)(arow + 32 * kDH + ko);
    }
    acc[0] = wmma_bf16(a0, bfr, acc[0]);
    acc[1] = wmma_bf16(a1, bfr, acc[1]);
    acc[2] = wmma_bf16(a2, bfr, acc[2]);
  }
  const int t = t0 + lo;
  if (t < kL) {
    long long dst = (t == 0) ? 0 : (1 + perm[t - 1]);
    float* obase = out + ((size_t)b * kL + dst) * kDM;
#pragma unroll
    for (int s = 0; s < 3; ++s) {
      float* orow = obase + e0 + s * 16 + 8 * hi;   // acc[s][r] -> ch +r
      float4 v0 = make_float4(acc[s][0], acc[s][1], acc[s][2], acc[s][3]);
      float4 v1 = make_float4(acc[s][4], acc[s][5], acc[s][6], acc[s][7]);
      *(float4*)(orow) = v0;
      *(float4*)(orow + 4) = v1;
    }
  }
}

// ---------------------------------------------------------------------------
// Launch
// ---------------------------------------------------------------------------
extern "C" void kernel_launch(void* const* d_in, const int* in_sizes, int n_in,
                              void* d_out, int out_size, void* d_ws, size_t ws_size,
                              hipStream_t stream) {
  (void)in_sizes; (void)n_in; (void)out_size; (void)ws_size;
  const float* hs    = (const float*)d_in[0];
  const float* w_in  = (const float*)d_in[1];
  const float* wcx   = (const float*)d_in[2];
  const float* wcz   = (const float*)d_in[3];
  const float* w_xp  = (const float*)d_in[4];
  const float* w_dt  = (const float*)d_in[5];
  const float* dtb   = (const float*)d_in[6];
  const float* Alog  = (const float*)d_in[7];
  const float* Dpar  = (const float*)d_in[8];
  const float* w_out = (const float*)d_in[9];
  const long long* perm = (const long long*)d_in[10];
  float* outp = (float*)d_out;

  char* base = (char*)d_ws;
  size_t off = 0;
  auto carve = [&](size_t bytes) -> void* {
    void* p = base + off;
    off += (bytes + 255) & ~(size_t)255;
    return p;
  };
  __bf16* hsbf = (__bf16*)carve((size_t)kB * kL * kDM * 2);
  __bf16* winb = (__bf16*)carve((size_t)kDI * kDM * 2);
  __bf16* wxpb = (__bf16*)carve((size_t)kNDBL * kDM * 2);
  __bf16* wdtb = (__bf16*)carve((size_t)kDH * 32 * 2);
  __bf16* woub = (__bf16*)carve((size_t)kDM * kDH * 2);
  float*  xz   = (float*) carve((size_t)kB * kDI * kL * 4);
  float*  xzc  = (float*) carve((size_t)kB * kDI * kL * 4);
  __bf16* xbf  = (__bf16*)carve((size_t)kB * kL * kDH * 2);
  float*  xdbl = (float*) carve((size_t)kB * kNDBL * kL * 4);
  float*  dtv  = (float*) carve((size_t)kB * kDH * kL * 4);
  __bf16* ybf  = (__bf16*)carve((size_t)kB * kL * kDH * 2);

  // --- weight conversions (bf16) ---
  cvt_bf16_kernel<<<(kDI * kDM + 255) / 256, 256, 0, stream>>>(w_in, winb, kDI * kDM);
  cvt_bf16_kernel<<<(kNDBL * kDM + 255) / 256, 256, 0, stream>>>(w_xp, wxpb, kNDBL * kDM);
  cvt_bf16_kernel<<<(kDM * kDH + 255) / 256, 256, 0, stream>>>(w_out, woub, kDM * kDH);
  cvt_dtw_kernel<<<(kDH * 32 + 255) / 256, 256, 0, stream>>>(w_dt, wdtb);

  // --- permuted bf16 activations ---
  {
    size_t total = (size_t)kB * kL * kDM;
    gather_hs_kernel<<<(unsigned)((total + 255) / 256), 256, 0, stream>>>(hs, perm, hsbf);
  }

  // --- in_proj (WMMA bf16, LDS-staged activations) ---
  gemm_in_proj_kernel<<<dim3(3, kTT, kB), 128, 0, stream>>>(hsbf, winb, xz);

  // --- depthwise conv + SiLU ---
  {
    size_t total = (size_t)kB * kDI * kL;
    conv_silu_kernel<<<(unsigned)((total + 255) / 256), 256, 0, stream>>>(xz, wcx, wcz, xzc, xbf);
  }

  // --- x_proj (WMMA bf16, register-blocked) ---
  gemm_x_proj_kernel<<<dim3(1, kTT, kB), 32, 0, stream>>>(xbf, wxpb, xdbl);

  // --- dt_proj (WMMA bf16, K padded 24->32) ---
  gemm_dt_proj_kernel<<<dim3(kDH / 16, kTT, kB), 32, 0, stream>>>(xdbl, wdtb, dtv);

  // --- selective scan (lane per (b,d,n), shfl reduction) ---
  scan_kernel<<<dim3(kDH / 16, kB), 256, 0, stream>>>(dtv, xzc, xdbl, Alog, dtb, Dpar, ybf);

  // --- out_proj (WMMA bf16, LDS-staged) + inverse-permutation scatter ---
  gemm_out_proj_kernel<<<dim3(kDM / 16 / 8, kTT, kB), 128, 0, stream>>>(ybf, woub, perm, outp);
}
